// MultiHeadAttention_15564961481361
// MI455X (gfx1250) — matrix operand order repaired and checked by reading, not compiled, not run on hardware
//
#include <hip/hip_runtime.h>
#include <hip/hip_bf16.h>

// ---------------------------------------------------------------------------
// MultiHeadAttention for MI455X (gfx1250), bf16 WMMA pipeline.
// B=4, N=M=2048, D_IN=1024, H=16, HS=64, D_OUT=1024.
//
//   1) cast f32 -> bf16; weights cast WITH transpose so every GEMM B-matrix
//      becomes col-major [col][k] and LDS staging is a contiguous b128 copy.
//   2) unified WMMA GEMM (32 rows/wave, 8 v_wmma per K-step); B tiles are
//      DMA'd into LDS with GLOBAL_LOAD_ASYNC_TO_LDS_B128, double-buffered
//      against the WMMA compute using s_wait_asynccnt.
//      V projection is stored head-transposed Vt[b][h][hs][m].
//   3) flash attention: online softmax in registers, QK^T and P*V on WMMA,
//      P transposed through 1KB/wave LDS.
//   4) same GEMM kernel with f32+bias epilogue for the output projection.
// ---------------------------------------------------------------------------

typedef __attribute__((ext_vector_type(16))) __bf16 v16bf;
typedef __attribute__((ext_vector_type(8)))  __bf16 v8bf;
typedef __attribute__((ext_vector_type(8)))  float  v8f;

#define D_IN   1024
#define HEADS  16
#define HS     64
#define NROWS  8192          // B*N = B*M

// workspace element offsets (bf16 elements)
#define OFF_XQ  ((size_t)0)
#define OFF_XK  ((size_t)8388608)
#define OFF_XV  ((size_t)16777216)
#define OFF_WQ  ((size_t)25165824)
#define OFF_WK  ((size_t)26214400)
#define OFF_WV  ((size_t)27262976)
#define OFF_WP  ((size_t)28311552)
#define OFF_Q   ((size_t)29360128)
#define OFF_K   ((size_t)37748736)
#define OFF_V   ((size_t)46137344)
#define OFF_CTX ((size_t)54525952)

__device__ __forceinline__ __bf16 f2bf(float f) {
  unsigned u = __float_as_uint(f);
  u += 0x7FFFu + ((u >> 16) & 1u);               // round to nearest even
  unsigned short s = (unsigned short)(u >> 16);
  return __builtin_bit_cast(__bf16, s);
}

__device__ __forceinline__ v16bf concat8(v8bf lo, v8bf hi) {
  return __builtin_shufflevector(lo, hi, 0,1,2,3,4,5,6,7,8,9,10,11,12,13,14,15);
}

// ---------------------------------------------------------------------------
// 1a) plain f32 -> bf16 cast (activations)
// ---------------------------------------------------------------------------
__global__ void cast_f32_bf16(const float* __restrict__ src,
                              __bf16* __restrict__ dst, int n) {
  int i = blockIdx.x * blockDim.x + threadIdx.x;
  int stride = gridDim.x * blockDim.x;
  for (; i < n; i += stride) dst[i] = f2bf(src[i]);
}

// 1b) per-head projection weights (H,D_IN,HS) -> transposed [h*64+o][i]
__global__ void cast_tr_head(const float* __restrict__ src,
                             __bf16* __restrict__ dst) {
  int idx = blockIdx.x * 256 + threadIdx.x;   // 1048576, i fastest (coalesced wr)
  int i = idx & 1023;
  int c = idx >> 10;                          // c = h*64 + o
  int h = c >> 6, o = c & 63;
  dst[idx] = f2bf(src[((size_t)h * 1024 + i) * 64 + o]);
}

// 1c) output projection weights (1024k x 1024c row-major) -> [c][k]
__global__ void cast_tr_flat(const float* __restrict__ src,
                             __bf16* __restrict__ dst) {
  int idx = blockIdx.x * 256 + threadIdx.x;   // 1048576, k fastest
  int k = idx & 1023;
  int c = idx >> 10;
  dst[idx] = f2bf(src[(size_t)k * 1024 + c]);
}

// ---------------------------------------------------------------------------
// 2/4) Unified WMMA GEMM:  out[row, cBase+col] = scale * A[row,:] . Wt[col,:]
//   A : 8192x1024 bf16 row-major; Wt : [1024 cols][1024 k] bf16 (col-major B)
//   grid = (32, 16); block = 256 (8 waves); block tile 256 rows x 64 cols;
//   wave tile 32x64 (2 A-frags, 8 accumulators, 8 WMMA per K-step).
//   B tiles double-buffered in LDS via async DMA (ASYNCcnt).
//   out_mode: 0 = bf16 [row][col]           (Q/K projections)
//             1 = bf16 Vt[(b*1024+col)][m]  (V projection, head-transposed)
//             2 = f32  [row][col] + bias    (output projection)
// ---------------------------------------------------------------------------
__global__ __launch_bounds__(256)
void wmma_gemm(const __bf16* __restrict__ A, const __bf16* __restrict__ Wt,
               void* __restrict__ outp, const float* __restrict__ bias,
               float scale, int out_mode) {
  // [2 buffers][col][k]; array sits at LDS offset 0 (only shared object)
  __shared__ __align__(16) __bf16 ldsB[2][64 * 32];
  const int lane = threadIdx.x & 31;
  const int wave = threadIdx.x >> 5;
  const int r    = lane & 15;
  const int hi   = lane >> 4;
  const int rowBase = blockIdx.x * 256 + wave * 32;
  const int cBase   = blockIdx.y * 64;

  v8f acc[2][4];
  v8f zero = {0.f,0.f,0.f,0.f,0.f,0.f,0.f,0.f};
#pragma unroll
  for (int g = 0; g < 2; ++g)
#pragma unroll
    for (int f = 0; f < 4; ++f) acc[g][f] = zero;

  const __bf16* Arow0 = A + (size_t)(rowBase + r) * D_IN;
  const __bf16* Arow1 = A + (size_t)(rowBase + 16 + r) * D_IN;

  // each thread copies one contiguous 16B chunk of the B tile per K-step
  const int scol = threadIdx.x >> 2;          // 0..63
  const int sck  = (threadIdx.x & 3) * 8;     // 0,8,16,24
  const __bf16* Bsrc = Wt + (size_t)(cBase + scol) * 1024 + sck;
  const unsigned dOff0 = (unsigned)((scol * 32 + sck) * sizeof(__bf16));

  // NOTE: &ldsB passed as (unused) asm operand so the LDS global is captured:
  // without it LLVM proves "never stored" and folds the B-frag loads to poison.

  // prologue: DMA first B tile into buffer 0
  asm volatile("global_load_async_to_lds_b128 %0, %1, off"
               :: "v"(dOff0), "v"(Bsrc), "v"(&ldsB[0][0]) : "memory");

  for (int k0 = 0; k0 < D_IN; k0 += 32) {
    const int cur = (k0 >> 5) & 1;
    if (k0 + 32 < D_IN) {
      // DMA next tile into the other buffer, then wait for current tile
      // (async loads complete in order: <=1 outstanding => current landed)
      const unsigned dOffN =
          (unsigned)(((cur ^ 1) * 2048 + scol * 32 + sck) * sizeof(__bf16));
      asm volatile("global_load_async_to_lds_b128 %0, %1, off"
                   :: "v"(dOffN), "v"(Bsrc + k0 + 32), "v"(&ldsB[0][0])
                   : "memory");
      asm volatile("s_wait_asynccnt 0x1" ::: "memory");
    } else {
      asm volatile("s_wait_asynccnt 0x0" ::: "memory");
    }
    __syncthreads();

    const __bf16* ap0 = Arow0 + k0 + 8 * hi;
    v16bf a0 = concat8(*(const v8bf*)ap0, *(const v8bf*)(ap0 + 16));
    const __bf16* ap1 = Arow1 + k0 + 8 * hi;
    v16bf a1 = concat8(*(const v8bf*)ap1, *(const v8bf*)(ap1 + 16));

    const __bf16* bbase = &ldsB[cur][0];
#pragma unroll
    for (int f = 0; f < 4; ++f) {
      const __bf16* bp = bbase + (f * 16 + r) * 32 + 8 * hi;
      v16bf bfrag = concat8(*(const v8bf*)bp, *(const v8bf*)(bp + 16));
      acc[0][f] = __builtin_amdgcn_wmma_f32_16x16x32_bf16(
          false, a0, false, bfrag, (short)0, acc[0][f], false, false);
      acc[1][f] = __builtin_amdgcn_wmma_f32_16x16x32_bf16(
          false, a1, false, bfrag, (short)0, acc[1][f], false, false);
    }
    __syncthreads();
  }

#pragma unroll
  for (int g = 0; g < 2; ++g) {
#pragma unroll
    for (int f = 0; f < 4; ++f) {
#pragma unroll
      for (int rr = 0; rr < 8; ++rr) {
        int row = rowBase + g * 16 + rr + 8 * hi;
        int col = cBase + f * 16 + r;
        float val = acc[g][f][rr] * scale;
        if (out_mode == 0) {
          ((__bf16*)outp)[(size_t)row * 1024 + col] = f2bf(val);
        } else if (out_mode == 1) {
          int b = row >> 11, mm = row & 2047;    // Vt[(b*1024+c)][m]
          ((__bf16*)outp)[((size_t)(b * 1024 + col)) * 2048 + mm] = f2bf(val);
        } else {
          ((float*)outp)[(size_t)row * 1024 + col] = val + bias[col];
        }
      }
    }
  }
}

// ---------------------------------------------------------------------------
// 3) Flash attention: one wave = 16 query rows of one (b,h).
//    grid = 1024 blocks x 256 threads (8 waves) = 8192 tiles = 4*16*128.
//    V is pre-transposed (Vt[b][h][hs][m]) so P*V B-frags are contiguous.
// ---------------------------------------------------------------------------
__global__ __launch_bounds__(256)
void attn_kernel(const __bf16* __restrict__ Q, const __bf16* __restrict__ K,
                 const __bf16* __restrict__ Vt, __bf16* __restrict__ Ctx) {
  __shared__ __align__(16) __bf16 lds[8 * 512];    // per wave: 16x32 P tile
  const int lane = threadIdx.x & 31;
  const int wave = threadIdx.x >> 5;
  __bf16* ldsP = lds + wave * 512;                 // [row*32 + col]

  const int tile = blockIdx.x * 8 + wave;          // 0..8191
  const int b  = tile >> 11;
  const int rm = tile & 2047;
  const int h  = rm >> 7;
  const int qt = rm & 127;
  const int r  = lane & 15;
  const int hi = lane >> 4;

  const __bf16* Qb  = Q  + ((size_t)(b * 2048 + qt * 16)) * 1024 + h * HS;
  const __bf16* Kb  = K  + (size_t)b * 2048 * 1024 + h * HS;
  const __bf16* Vtb = Vt + ((size_t)(b * 1024 + h * HS)) * 2048;  // [hs][m]

  // Q A-fragments for K-dim 0..31 and 32..63 (live in registers all loop)
  v16bf qa[2];
#pragma unroll
  for (int ks = 0; ks < 2; ++ks) {
    const __bf16* qp = Qb + (size_t)r * 1024 + ks * 32 + 8 * hi;
    qa[ks] = concat8(*(const v8bf*)qp, *(const v8bf*)(qp + 16));
  }

  float m[8], l[8];
  v8f o[4];
  v8f zero = {0.f,0.f,0.f,0.f,0.f,0.f,0.f,0.f};
#pragma unroll
  for (int f = 0; f < 4; ++f) o[f] = zero;
#pragma unroll
  for (int rr = 0; rr < 8; ++rr) { m[rr] = -1e30f; l[rr] = 0.f; }

  for (int kb = 0; kb < 2048; kb += 32) {
    // S = Q * K^T for 32 keys (two 16-key column tiles)
    v8f s0 = zero, s1 = zero;
#pragma unroll
    for (int ks = 0; ks < 2; ++ks) {
      const __bf16* kp0 = Kb + (size_t)(kb + r) * 1024 + ks * 32 + 8 * hi;
      v16bf b0 = concat8(*(const v8bf*)kp0, *(const v8bf*)(kp0 + 16));
      s0 = __builtin_amdgcn_wmma_f32_16x16x32_bf16(
          false, qa[ks], false, b0, (short)0, s0, false, false);
      const __bf16* kp1 = Kb + (size_t)(kb + 16 + r) * 1024 + ks * 32 + 8 * hi;
      v16bf b1 = concat8(*(const v8bf*)kp1, *(const v8bf*)(kp1 + 16));
      s1 = __builtin_amdgcn_wmma_f32_16x16x32_bf16(
          false, qa[ks], false, b1, (short)0, s1, false, false);
    }

    // online softmax update (rows in regs rr, cols across 16-lane halves)
#pragma unroll
    for (int rr = 0; rr < 8; ++rr) {
      float smax = fmaxf(s0[rr], s1[rr]);
#pragma unroll
      for (int mk = 1; mk <= 8; mk <<= 1)
        smax = fmaxf(smax, __shfl_xor(smax, mk, 32));
      float mnew = fmaxf(m[rr], smax);
      float corr = __expf(m[rr] - mnew);
      float p0 = __expf(s0[rr] - mnew);
      float p1 = __expf(s1[rr] - mnew);
      float rs = p0 + p1;
#pragma unroll
      for (int mk = 1; mk <= 8; mk <<= 1) rs += __shfl_xor(rs, mk, 32);
      l[rr] = l[rr] * corr + rs;
      m[rr] = mnew;
#pragma unroll
      for (int f = 0; f < 4; ++f) o[f][rr] *= corr;
      int prow = rr + 8 * hi;
      ldsP[prow * 32 + r]      = f2bf(p0);
      ldsP[prow * 32 + 16 + r] = f2bf(p1);
    }

    // transpose P through LDS into A-frag layout, then O += P * V
    const __bf16* pp = ldsP + r * 32 + 8 * hi;
    v16bf pa = concat8(*(const v8bf*)pp, *(const v8bf*)(pp + 16));
#pragma unroll
    for (int f = 0; f < 4; ++f) {
      const __bf16* vp = Vtb + (size_t)(f * 16 + r) * 2048 + kb + 8 * hi;
      v16bf vb = concat8(*(const v8bf*)vp, *(const v8bf*)(vp + 16));
      o[f] = __builtin_amdgcn_wmma_f32_16x16x32_bf16(
          false, pa, false, vb, (short)0, o[f], false, false);
    }
  }

  // normalize and store context (bf16)
#pragma unroll
  for (int rr = 0; rr < 8; ++rr) {
    float inv = 1.0f / l[rr];
    int row = b * 2048 + qt * 16 + rr + 8 * hi;
#pragma unroll
    for (int f = 0; f < 4; ++f) {
      int col = h * HS + f * 16 + r;
      Ctx[(size_t)row * 1024 + col] = f2bf(o[f][rr] * inv);
    }
  }
}

// ---------------------------------------------------------------------------
extern "C" void kernel_launch(void* const* d_in, const int* in_sizes, int n_in,
                              void* d_out, int out_size, void* d_ws, size_t ws_size,
                              hipStream_t stream) {
  const float* query = (const float*)d_in[0];
  const float* key   = (const float*)d_in[1];
  const float* value = (const float*)d_in[2];
  const float* wq    = (const float*)d_in[3];
  const float* wk    = (const float*)d_in[4];
  const float* wv    = (const float*)d_in[5];
  const float* wp    = (const float*)d_in[6];
  const float* pbias = (const float*)d_in[7];
  float* out = (float*)d_out;
  __bf16* ws = (__bf16*)d_ws;

  const int nBig = 4 * 2048 * 1024;   // 8388608

  // 1) casts (weights transposed to col-major [col][k])
  cast_f32_bf16<<<1024, 256, 0, stream>>>(query, ws + OFF_XQ, nBig);
  cast_f32_bf16<<<1024, 256, 0, stream>>>(key,   ws + OFF_XK, nBig);
  cast_f32_bf16<<<1024, 256, 0, stream>>>(value, ws + OFF_XV, nBig);
  cast_tr_head<<<4096, 256, 0, stream>>>(wq, ws + OFF_WQ);
  cast_tr_head<<<4096, 256, 0, stream>>>(wk, ws + OFF_WK);
  cast_tr_head<<<4096, 256, 0, stream>>>(wv, ws + OFF_WV);
  cast_tr_flat<<<4096, 256, 0, stream>>>(wp, ws + OFF_WP);

  // 2) projections (1/sqrt(64) folded into Q; V stored head-transposed)
  dim3 ggrid(NROWS / 256, HEADS);
  wmma_gemm<<<ggrid, 256, 0, stream>>>(ws + OFF_XQ, ws + OFF_WQ, ws + OFF_Q,
                                       nullptr, 0.125f, 0);
  wmma_gemm<<<ggrid, 256, 0, stream>>>(ws + OFF_XK, ws + OFF_WK, ws + OFF_K,
                                       nullptr, 1.0f, 0);
  wmma_gemm<<<ggrid, 256, 0, stream>>>(ws + OFF_XV, ws + OFF_WV, ws + OFF_V,
                                       nullptr, 1.0f, 1);

  // 3) attention (8192 wave-tiles)
  attn_kernel<<<1024, 256, 0, stream>>>(ws + OFF_Q, ws + OFF_K, ws + OFF_V,
                                        ws + OFF_CTX);

  // 4) output projection + bias (f32 out)
  wmma_gemm<<<ggrid, 256, 0, stream>>>(ws + OFF_CTX, ws + OFF_WP, out,
                                       pbias, 1.0f, 2);
}
